// NeuroSAT_49830210568545
// MI455X (gfx1250) — matrix operand order repaired and checked by reading, not compiled, not run on hardware
//
#include <hip/hip_runtime.h>
#include <hip/hip_bf16.h>
#include <math.h>
#include <stdint.h>

typedef __bf16 bf16_t;
typedef __attribute__((ext_vector_type(16))) __bf16 v16bf;
typedef __attribute__((ext_vector_type(8)))  __bf16 v8bf;
typedef __attribute__((ext_vector_type(8)))  float   v8f;

#define NS_DIM 128

// ---------------------------------------------------------------------------
// Tensor Data Mover availability (ROCm 7.2 clang-22: 5 args; clang-23+: 6 args)
// ---------------------------------------------------------------------------
#if defined(__has_builtin)
#  if __has_builtin(__builtin_amdgcn_tensor_load_to_lds)
#    define NS_TDM 1
#  endif
#endif
#ifndef NS_TDM
#  define NS_TDM 0
#endif

constexpr int BM = 128;
constexpr int BN = 128;
constexpr int BK = 32;
constexpr int LDSW = 40;  // padded LDS row stride in bf16 (80B): TDM pad 4dw / 16dw

#if NS_TDM
typedef __attribute__((ext_vector_type(4))) unsigned int u32x4;
typedef __attribute__((ext_vector_type(4))) int          i32x4;
typedef __attribute__((ext_vector_type(8))) int          i32x8;

// Issue one TDM 2D tile load: [heightRows x widthElems clipped] of bf16 from
// gsrc (row stride ldElems) into LDS at lds_off, tile 128 x 32 elements, with
// LDS padding of 4 DWORDs every 16 DWORDs (-> 80B rows). OOB rows zero-fill.
__device__ __forceinline__ void tdm_load_tile(const bf16_t* gsrc, unsigned lds_off,
                                              int widthElems, int heightRows,
                                              int ldElems) {
  const unsigned long long ga = (unsigned long long)(uintptr_t)gsrc;
  u32x4 g0;
  g0[0] = 1u;                                                 // count=1, user D#
  g0[1] = lds_off;                                            // lds_addr (bytes)
  g0[2] = (unsigned)ga;                                       // global_addr[31:0]
  g0[3] = (unsigned)((ga >> 32) & 0x01ffffffu) | (2u << 30);  // addr[56:32]|type=2
  const unsigned d0 = (unsigned)widthElems;
  const unsigned d1 = (unsigned)(heightRows > 0 ? heightRows : 0);
  const unsigned s0 = (unsigned)ldElems;
  i32x8 g1;
  // data_size=1 (2B) @16 | pad_enable @20 | pad_interval=3 (16dw) @22 | pad_amount=3 (4dw) @25
  g1[0] = (int)((1u << 16) | (1u << 20) | (3u << 22) | (3u << 25));
  g1[1] = (int)(d0 << 16);                            // tensor_dim0[15:0]
  g1[2] = (int)((d0 >> 16) | (d1 << 16));             // dim0[31:16] | dim1[15:0]
  g1[3] = (int)((d1 >> 16) | ((unsigned)BK << 16));   // dim1[31:16] | tile_dim0=32
  g1[4] = (int)(unsigned)BM;                          // tile_dim1=128, tile_dim2=0
  g1[5] = (int)s0;                                    // tensor_dim0_stride[31:0]
  g1[6] = 0;                                          // stride[47:32] | dim1_stride lo
  g1[7] = 0;
  const i32x4 z4 = {0, 0, 0, 0};
#if __clang_major__ >= 23
  const i32x8 z8 = {0, 0, 0, 0, 0, 0, 0, 0};
  __builtin_amdgcn_tensor_load_to_lds(g0, g1, z4, z4, z8, 0);
#else
  __builtin_amdgcn_tensor_load_to_lds(g0, g1, z4, z4, 0);
#endif
}
#endif  // NS_TDM

union ABfrag { v16bf v; v8bf h[2]; };

// One 16-row strip x 8 N-tiles of WMMA from staged LDS tiles.
__device__ __forceinline__ void ns_compute_tiles(const bf16_t* __restrict__ Xs,
                                                 const bf16_t* __restrict__ Ws,
                                                 int wave, int lane, v8f (&acc)[8]) {
  // A fragment (ISA 16-bit A 16x32): lane<16 -> K{0..7,16..23}, lane>=16 -> +8
  const int r  = (wave << 4) + (lane & 15);
  const int cb = (lane >> 4) << 3;  // 0 or 8
  ABfrag a;
  a.h[0] = *(const v8bf*)&Xs[r * LDSW + cb];
  a.h[1] = *(const v8bf*)&Xs[r * LDSW + 16 + cb];
  // B fragment: same K order as A: lane<16 -> K{0..7,16..23}, lane>=16 -> +8
  const int koffB = (lane >> 4) << 3;  // 0 or 8
  #pragma unroll
  for (int nt = 0; nt < 8; ++nt) {
    const int n = (nt << 4) + (lane & 15);
    ABfrag b;
    b.h[0] = *(const v8bf*)&Ws[n * LDSW + koffB];
    b.h[1] = *(const v8bf*)&Ws[n * LDSW + koffB + 16];
    acc[nt] = __builtin_amdgcn_wmma_f32_16x16x32_bf16(
        false, a.v, false, b.v, (short)0, acc[nt], false, false);
  }
}

// ---------------------------------------------------------------------------
// WMMA GEMM: Y[M x N] = act(X[M x K] @ W[N x K]^T + bias)
//   X: bf16 row-major ld=K; W: bf16 row-major (N x K)
//   OUT: 0 = store bf16, 1 = store f32, 2 = accumulate into f32 (bias ignored)
// ---------------------------------------------------------------------------
template<int OUT, bool RELU>
__global__ __launch_bounds__(256)
void gemm_kernel(const bf16_t* __restrict__ X,
                 const bf16_t* __restrict__ W,
                 const float* __restrict__ bias,
                 void* __restrict__ Yv,
                 int M, int N, int K)
{
  __shared__ __align__(16) bf16_t Xs[2][BM * LDSW];
  __shared__ __align__(16) bf16_t Ws[2][BN * LDSW];

  const int tid     = threadIdx.x;
  const int lane    = tid & 31;
  const int wave    = tid >> 5;
  const int rowBase = blockIdx.x * BM;
  const int colBase = blockIdx.y * BN;
  const int nk      = K / BK;

  v8f acc[8] = {};

#if NS_TDM
  // --- double-buffered TDM pipeline: wave 0 drives the Tensor Data Mover ---
  if (wave == 0) {
    tdm_load_tile(X + (size_t)rowBase * K, (unsigned)(uintptr_t)&Xs[0][0],
                  K, M - rowBase, K);
    tdm_load_tile(W + (size_t)colBase * K, (unsigned)(uintptr_t)&Ws[0][0],
                  K, BN, K);
  }
  for (int kt = 0; kt < nk; ++kt) {
    const int buf = kt & 1;
    if (wave == 0) {
      if (kt + 1 < nk) {
        const int kOff2 = (kt + 1) * BK;
        tdm_load_tile(X + (size_t)rowBase * K + kOff2,
                      (unsigned)(uintptr_t)&Xs[buf ^ 1][0],
                      K - kOff2, M - rowBase, K);
        tdm_load_tile(W + (size_t)colBase * K + kOff2,
                      (unsigned)(uintptr_t)&Ws[buf ^ 1][0],
                      K - kOff2, BN, K);
        __builtin_amdgcn_s_wait_tensorcnt(2);  // current buffer's 2 loads done
      } else {
        __builtin_amdgcn_s_wait_tensorcnt(0);
      }
    }
    __syncthreads();                      // release waves: tile[buf] is ready
    ns_compute_tiles(&Xs[buf][0], &Ws[buf][0], wave, lane, acc);
    __syncthreads();                      // compute done before buf is re-filled
  }
#else
  // --- fallback: cooperative per-thread staging ---
  const int ldRow  = tid >> 1;        // 0..127
  const int ldHalf = (tid & 1) << 4;  // 0 or 16 elements within 32-wide K chunk
  for (int kt = 0; kt < nk; ++kt) {
    const int kOff = kt * BK;
    __syncthreads();
    {
      bf16_t* dst = &Xs[0][ldRow * LDSW + ldHalf];
      const int g = rowBase + ldRow;
      if (g < M) {
        const v8bf* p = (const v8bf*)(X + (size_t)g * K + kOff + ldHalf);
        *(v8bf*)(dst)     = p[0];
        *(v8bf*)(dst + 8) = p[1];
      } else {
        #pragma unroll
        for (int q = 0; q < 16; ++q) dst[q] = (bf16_t)0.0f;
      }
      bf16_t* dw = &Ws[0][ldRow * LDSW + ldHalf];
      const v8bf* pw = (const v8bf*)(W + (size_t)(colBase + ldRow) * K + kOff + ldHalf);
      *(v8bf*)(dw)     = pw[0];
      *(v8bf*)(dw + 8) = pw[1];
    }
    __syncthreads();
    ns_compute_tiles(&Xs[0][0], &Ws[0][0], wave, lane, acc);
  }
#endif

  // ---- epilogue: C/D layout VGPR j -> M = j + 8*(lane>=16), N = lane&15 ----
  const int mBase = rowBase + (wave << 4) + ((lane >> 4) << 3);
  #pragma unroll
  for (int nt = 0; nt < 8; ++nt) {
    const int col = colBase + (nt << 4) + (lane & 15);
    const float bv = bias ? bias[col] : 0.0f;
    #pragma unroll
    for (int j = 0; j < 8; ++j) {
      const int m = mBase + j;
      if (m < M) {
        float v = acc[nt][j] + bv;
        if (RELU) v = v > 0.0f ? v : 0.0f;
        if constexpr (OUT == 0)
          ((bf16_t*)Yv)[(size_t)m * N + col] = (bf16_t)v;
        else if constexpr (OUT == 1)
          ((float*)Yv)[(size_t)m * N + col] = v;
        else
          ((float*)Yv)[(size_t)m * N + col] += v;
      }
    }
  }
}

// ---------------------------------------------------------------------------
// Support kernels
// ---------------------------------------------------------------------------
__global__ void k_f32_to_bf16(const float* __restrict__ s, bf16_t* __restrict__ d, int n) {
  int i = blockIdx.x * 256 + threadIdx.x;
  if (i < n) d[i] = (bf16_t)s[i];
}

__global__ void k_zero_f32(float* __restrict__ p, size_t n) {
  size_t i = (size_t)blockIdx.x * 256 + threadIdx.x;
  size_t stride = (size_t)gridDim.x * 256;
  for (; i < n; i += stride) p[i] = 0.0f;
}

__global__ void k_add_vec(const float* a, const float* b, float* o, int n) {
  int i = blockIdx.x * 256 + threadIdx.x;
  if (i < n) o[i] = a[i] + b[i];
}

// h[m, d] = bf16(w[d] + b[d])   (Linear(1,dim) applied to ts=1.0)
__global__ void k_init_h(const float* __restrict__ w, const float* __restrict__ b,
                         bf16_t* __restrict__ h, int total) {
  int i = blockIdx.x * 256 + threadIdx.x;
  if (i < total) { int d = i & (NS_DIM - 1); h[i] = (bf16_t)(w[d] + b[d]); }
}

// segment-sum scatter: dst[sidx[e], :] += src[gidx[e], :] ; one wave per edge
__global__ __launch_bounds__(256)
void k_scatter(const bf16_t* __restrict__ src, const int* __restrict__ gidx,
               const int* __restrict__ sidx, float* __restrict__ dst, int n_edges) {
  int e = blockIdx.x * 8 + (threadIdx.x >> 5);
  if (e >= n_edges) return;
  const int lane = threadIdx.x & 31;
  const bf16_t* sp = src + (size_t)gidx[e] * NS_DIM;
  float*        dp = dst + (size_t)sidx[e] * NS_DIM;
  #pragma unroll
  for (int j = 0; j < 4; ++j)
    atomicAdd(&dp[lane + 32 * j], (float)sp[lane + 32 * j]);
}

__device__ __forceinline__ float ns_sigmoid(float x) { return 1.0f / (1.0f + __expf(-x)); }

// z[m, 0:512] = (i,f,g,o) pre-activations; update c (f32) and h (bf16)
__global__ void k_lstm_gates(const float* __restrict__ z, float* __restrict__ c,
                             bf16_t* __restrict__ h, int M) {
  int idx = blockIdx.x * 256 + threadIdx.x;
  if (idx >= M * NS_DIM) return;
  const int m = idx >> 7, d = idx & (NS_DIM - 1);
  const float* zp = z + (size_t)m * 512;
  float ig = ns_sigmoid(zp[d]);
  float fg = ns_sigmoid(zp[NS_DIM + d]);
  float gg = tanhf(zp[2 * NS_DIM + d]);
  float og = ns_sigmoid(zp[3 * NS_DIM + d]);
  float cn = fg * c[idx] + ig * gg;
  c[idx] = cn;
  h[idx] = (bf16_t)(og * tanhf(cn));
}

// Xin[m, 0:128] = bf16(CL[m]); Xin[m, 128:256] = L_h[flip(m)]
__global__ void k_build_lstm_in(const float* __restrict__ CL, const bf16_t* __restrict__ Lh,
                                bf16_t* __restrict__ Xin, int n_vars, int n_lits) {
  int idx = blockIdx.x * 256 + threadIdx.x;
  if (idx >= n_lits * NS_DIM) return;
  const int m = idx >> 7, d = idx & (NS_DIM - 1);
  const int fm = (m < n_vars) ? m + n_vars : m - n_vars;
  bf16_t* xp = Xin + (size_t)m * (2 * NS_DIM);
  xp[d]          = (bf16_t)CL[idx];
  xp[NS_DIM + d] = Lh[(size_t)fm * NS_DIM + d];
}

// final vote layer: vote[m] = dot(h[m,:], w3) + b3 ; one wave per row
__global__ __launch_bounds__(256)
void k_vote(const bf16_t* __restrict__ h, const bf16_t* __restrict__ w,
            const float* __restrict__ b3, float* __restrict__ vote, int M) {
  int m = blockIdx.x * 8 + (threadIdx.x >> 5);
  if (m >= M) return;
  const int lane = threadIdx.x & 31;
  const bf16_t* hp = h + (size_t)m * NS_DIM;
  float s = 0.0f;
  #pragma unroll
  for (int j = 0; j < 4; ++j)
    s += (float)hp[lane + 32 * j] * (float)w[lane + 32 * j];
  #pragma unroll
  for (int off = 16; off > 0; off >>= 1) s += __shfl_down(s, off);
  if (lane == 0) vote[m] = s + b3[0];
}

// out[p] = mean over 2*per values (vote[v], vote[v+n_vars]) for v in prob p
__global__ void k_mean(const float* __restrict__ vote, float* __restrict__ out,
                       int n_vars, int n_probs) {
  __shared__ float sh[256];
  const int p = blockIdx.x;
  const int per = n_vars / n_probs;
  float s = 0.0f;
  for (int v = threadIdx.x; v < per; v += 256) {
    int g = p * per + v;
    s += vote[g] + vote[g + n_vars];
  }
  sh[threadIdx.x] = s;
  __syncthreads();
  for (int w = 128; w > 0; w >>= 1) {
    if (threadIdx.x < w) sh[threadIdx.x] += sh[threadIdx.x + w];
    __syncthreads();
  }
  if (threadIdx.x == 0) out[p] = sh[0] / (2.0f * per);
}

// ---------------------------------------------------------------------------
// Host orchestration
// ---------------------------------------------------------------------------
extern "C" void kernel_launch(void* const* d_in, const int* in_sizes, int n_in,
                              void* d_out, int out_size, void* d_ws, size_t ws_size,
                              hipStream_t stream) {
  (void)n_in; (void)out_size; (void)ws_size;

  const int n_vars = 50000, n_clauses = 210000, n_probs = 50, n_rounds = 4;
  const int n_lits = 2 * n_vars;
  const int n_cells = in_sizes[0];

  const int*   edge_lits    = (const int*)d_in[0];
  const int*   edge_clauses = (const int*)d_in[1];
  const float* L_init_w = (const float*)d_in[6];
  const float* L_init_b = (const float*)d_in[7];
  const float* C_init_w = (const float*)d_in[8];
  const float* C_init_b = (const float*)d_in[9];
  const float* Lmsg_w[3] = {(const float*)d_in[10], (const float*)d_in[12], (const float*)d_in[14]};
  const float* Lmsg_b[3] = {(const float*)d_in[11], (const float*)d_in[13], (const float*)d_in[15]};
  const float* Cmsg_w[3] = {(const float*)d_in[16], (const float*)d_in[18], (const float*)d_in[20]};
  const float* Cmsg_b[3] = {(const float*)d_in[17], (const float*)d_in[19], (const float*)d_in[21]};
  const float* Lvote_w1 = (const float*)d_in[22];
  const float* Lvote_b1 = (const float*)d_in[23];
  const float* Lvote_w2 = (const float*)d_in[24];
  const float* Lvote_b2 = (const float*)d_in[25];
  const float* Lvote_w3 = (const float*)d_in[26];
  const float* Lvote_b3 = (const float*)d_in[27];
  const float* Lu_wih = (const float*)d_in[28];
  const float* Lu_whh = (const float*)d_in[29];
  const float* Lu_bih = (const float*)d_in[30];
  const float* Lu_bhh = (const float*)d_in[31];
  const float* Cu_wih = (const float*)d_in[32];
  const float* Cu_whh = (const float*)d_in[33];
  const float* Cu_bih = (const float*)d_in[34];
  const float* Cu_bhh = (const float*)d_in[35];

  // ---- workspace carve (bump allocator, 256B aligned) ----
  char* wp = (char*)d_ws;
  auto alloc = [&](size_t bytes) -> void* {
    void* r = wp; wp += (bytes + 255) & ~(size_t)255; return r;
  };
  bf16_t* LwB[3]; bf16_t* CwB[3]; bf16_t* VwB[2];
  for (int i = 0; i < 3; ++i) LwB[i] = (bf16_t*)alloc(16384 * sizeof(bf16_t));
  for (int i = 0; i < 3; ++i) CwB[i] = (bf16_t*)alloc(16384 * sizeof(bf16_t));
  for (int i = 0; i < 2; ++i) VwB[i] = (bf16_t*)alloc(16384 * sizeof(bf16_t));
  bf16_t* Vw3B   = (bf16_t*)alloc(128 * sizeof(bf16_t));
  bf16_t* LuWihB = (bf16_t*)alloc((size_t)512 * 256 * sizeof(bf16_t));
  bf16_t* LuWhhB = (bf16_t*)alloc((size_t)512 * 128 * sizeof(bf16_t));
  bf16_t* CuWihB = (bf16_t*)alloc((size_t)512 * 128 * sizeof(bf16_t));
  bf16_t* CuWhhB = (bf16_t*)alloc((size_t)512 * 128 * sizeof(bf16_t));
  float*  Lbsum  = (float*)alloc(512 * sizeof(float));
  float*  Cbsum  = (float*)alloc(512 * sizeof(float));
  bf16_t* L_h  = (bf16_t*)alloc((size_t)n_lits    * NS_DIM * sizeof(bf16_t));
  bf16_t* C_h  = (bf16_t*)alloc((size_t)n_clauses * NS_DIM * sizeof(bf16_t));
  float*  L_c  = (float*) alloc((size_t)n_lits    * NS_DIM * sizeof(float));
  float*  C_c  = (float*) alloc((size_t)n_clauses * NS_DIM * sizeof(float));
  bf16_t* tmpA = (bf16_t*)alloc((size_t)n_clauses * NS_DIM * sizeof(bf16_t));
  bf16_t* tmpB = (bf16_t*)alloc((size_t)n_clauses * NS_DIM * sizeof(bf16_t));
  float*  seg  = (float*) alloc((size_t)n_clauses * NS_DIM * sizeof(float));
  float*  z    = (float*) alloc((size_t)n_clauses * 512    * sizeof(float));
  bf16_t* Xin  = (bf16_t*)alloc((size_t)n_lits * 2 * NS_DIM * sizeof(bf16_t));
  float*  vote = (float*) alloc((size_t)n_lits * sizeof(float));

  auto cvt = [&](const float* s, bf16_t* d, int n) {
    k_f32_to_bf16<<<(n + 255) / 256, 256, 0, stream>>>(s, d, n);
  };
  for (int i = 0; i < 3; ++i) cvt(Lmsg_w[i], LwB[i], 16384);
  for (int i = 0; i < 3; ++i) cvt(Cmsg_w[i], CwB[i], 16384);
  cvt(Lvote_w1, VwB[0], 16384);
  cvt(Lvote_w2, VwB[1], 16384);
  cvt(Lvote_w3, Vw3B, 128);
  cvt(Lu_wih, LuWihB, 512 * 256);
  cvt(Lu_whh, LuWhhB, 512 * 128);
  cvt(Cu_wih, CuWihB, 512 * 128);
  cvt(Cu_whh, CuWhhB, 512 * 128);
  k_add_vec<<<2, 256, 0, stream>>>(Lu_bih, Lu_bhh, Lbsum, 512);
  k_add_vec<<<2, 256, 0, stream>>>(Cu_bih, Cu_bhh, Cbsum, 512);

  // ---- initial state ----
  k_init_h<<<(n_lits * NS_DIM + 255) / 256, 256, 0, stream>>>(L_init_w, L_init_b, L_h, n_lits * NS_DIM);
  k_init_h<<<(n_clauses * NS_DIM + 255) / 256, 256, 0, stream>>>(C_init_w, C_init_b, C_h, n_clauses * NS_DIM);
  k_zero_f32<<<4096, 256, 0, stream>>>(L_c, (size_t)n_lits * NS_DIM);
  k_zero_f32<<<4096, 256, 0, stream>>>(C_c, (size_t)n_clauses * NS_DIM);

  const dim3 blk(256);
  auto grid128 = [](int M, int N) { return dim3((M + BM - 1) / BM, N / BN); };

  for (int r = 0; r < n_rounds; ++r) {
    // L_pre = MLP(L_h)
    gemm_kernel<0, true ><<<grid128(n_lits, 128), blk, 0, stream>>>(L_h,  LwB[0], Lmsg_b[0], tmpA, n_lits, 128, 128);
    gemm_kernel<0, true ><<<grid128(n_lits, 128), blk, 0, stream>>>(tmpA, LwB[1], Lmsg_b[1], tmpB, n_lits, 128, 128);
    gemm_kernel<0, false><<<grid128(n_lits, 128), blk, 0, stream>>>(tmpB, LwB[2], Lmsg_b[2], tmpA, n_lits, 128, 128);
    // LC = segment_sum(L_pre[edge_lits] -> edge_clauses)
    k_zero_f32<<<4096, 256, 0, stream>>>(seg, (size_t)n_clauses * NS_DIM);
    k_scatter<<<(n_cells + 7) / 8, blk, 0, stream>>>(tmpA, edge_lits, edge_clauses, seg, n_cells);
    // C LSTM: z = LC@Wih^T + C_h@Whh^T + (bih+bhh)
    k_f32_to_bf16<<<(n_clauses * NS_DIM + 255) / 256, blk, 0, stream>>>(seg, tmpB, n_clauses * NS_DIM);
    gemm_kernel<1, false><<<grid128(n_clauses, 512), blk, 0, stream>>>(tmpB, CuWihB, Cbsum,   z, n_clauses, 512, 128);
    gemm_kernel<2, false><<<grid128(n_clauses, 512), blk, 0, stream>>>(C_h,  CuWhhB, nullptr, z, n_clauses, 512, 128);
    k_lstm_gates<<<(n_clauses * NS_DIM + 255) / 256, blk, 0, stream>>>(z, C_c, C_h, n_clauses);
    // C_pre = MLP(C_h)
    gemm_kernel<0, true ><<<grid128(n_clauses, 128), blk, 0, stream>>>(C_h,  CwB[0], Cmsg_b[0], tmpA, n_clauses, 128, 128);
    gemm_kernel<0, true ><<<grid128(n_clauses, 128), blk, 0, stream>>>(tmpA, CwB[1], Cmsg_b[1], tmpB, n_clauses, 128, 128);
    gemm_kernel<0, false><<<grid128(n_clauses, 128), blk, 0, stream>>>(tmpB, CwB[2], Cmsg_b[2], tmpA, n_clauses, 128, 128);
    // CL = segment_sum(C_pre[edge_clauses] -> edge_lits)
    k_zero_f32<<<4096, 256, 0, stream>>>(seg, (size_t)n_lits * NS_DIM);
    k_scatter<<<(n_cells + 7) / 8, blk, 0, stream>>>(tmpA, edge_clauses, edge_lits, seg, n_cells);
    // L LSTM on concat(CL, flipped)
    k_build_lstm_in<<<(n_lits * NS_DIM + 255) / 256, blk, 0, stream>>>(seg, L_h, Xin, n_vars, n_lits);
    gemm_kernel<1, false><<<grid128(n_lits, 512), blk, 0, stream>>>(Xin, LuWihB, Lbsum,   z, n_lits, 512, 256);
    gemm_kernel<2, false><<<grid128(n_lits, 512), blk, 0, stream>>>(L_h, LuWhhB, nullptr, z, n_lits, 512, 128);
    k_lstm_gates<<<(n_lits * NS_DIM + 255) / 256, blk, 0, stream>>>(z, L_c, L_h, n_lits);
  }

  // vote MLP + mean
  gemm_kernel<0, true ><<<grid128(n_lits, 128), blk, 0, stream>>>(L_h,  VwB[0], Lvote_b1, tmpA, n_lits, 128, 128);
  gemm_kernel<0, true ><<<grid128(n_lits, 128), blk, 0, stream>>>(tmpA, VwB[1], Lvote_b2, tmpB, n_lits, 128, 128);
  k_vote<<<(n_lits + 7) / 8, blk, 0, stream>>>(tmpB, Vw3B, Lvote_b3, vote, n_lits);
  k_mean<<<n_probs, blk, 0, stream>>>(vote, (float*)d_out, n_vars, n_probs);
}
